// PathwayCoherenceLoss_66838281060554
// MI455X (gfx1250) — matrix operand order repaired and checked by reading, compiled, not run on hardware
//
#include <hip/hip_runtime.h>

// Problem constants (match the reference harness)
#define BATCH   256
#define NGENES  20000
#define NPATH   2000

typedef __attribute__((ext_vector_type(16))) __bf16 v16bf;
typedef __attribute__((ext_vector_type(8)))  __bf16 v8bf;
typedef __attribute__((ext_vector_type(8)))  float  v8f;

// ---------------------------------------------------------------------------
// Kernel 0: Dbf = bf16(pred - expr), row-major (BATCH, NGENES)
// ---------------------------------------------------------------------------
__global__ __launch_bounds__(256) void pcl_diff(const float* __restrict__ e,
                                                const float* __restrict__ p,
                                                __bf16* __restrict__ Dbf) {
    size_t i = ((size_t)blockIdx.x * 256 + threadIdx.x) * 8;
    float4 e0 = *(const float4*)(e + i);
    float4 e1 = *(const float4*)(e + i + 4);
    float4 p0 = *(const float4*)(p + i);
    float4 p1 = *(const float4*)(p + i + 4);
    v8bf o;
    o[0] = (__bf16)(p0.x - e0.x); o[1] = (__bf16)(p0.y - e0.y);
    o[2] = (__bf16)(p0.z - e0.z); o[3] = (__bf16)(p0.w - e0.w);
    o[4] = (__bf16)(p1.x - e1.x); o[5] = (__bf16)(p1.y - e1.y);
    o[6] = (__bf16)(p1.z - e1.z); o[7] = (__bf16)(p1.w - e1.w);
    *(uint4*)(Dbf + i) = __builtin_bit_cast(uint4, o);
}

// ---------------------------------------------------------------------------
// Kernel 1a: sizes only (small-workspace fallback)
// ---------------------------------------------------------------------------
__global__ __launch_bounds__(256) void pcl_sizes(const float* __restrict__ W,
                                                 float* __restrict__ invsz,
                                                 float* __restrict__ validf) {
    const int p = blockIdx.x;
    __shared__ float red[256];
    float s = 0.0f;
    const float* row = W + (size_t)p * NGENES;
    for (int g = threadIdx.x * 8; g < NGENES; g += 256 * 8) {
        float4 a = *(const float4*)(row + g);
        float4 b = *(const float4*)(row + g + 4);
        s += (a.x + a.y) + (a.z + a.w) + (b.x + b.y) + (b.z + b.w);
    }
    red[threadIdx.x] = s;
    __syncthreads();
    for (int off = 128; off > 0; off >>= 1) {
        if (threadIdx.x < off) red[threadIdx.x] += red[threadIdx.x + off];
        __syncthreads();
    }
    if (threadIdx.x == 0) {
        float sz = red[0];
        bool  v  = (sz >= 5.0f);
        float ss = fmaxf(sz, 1.0f);
        validf[p] = v ? 1.0f : 0.0f;
        invsz[p]  = v ? 1.0f / (ss * ss) : 0.0f;
    }
}

// ---------------------------------------------------------------------------
// Kernel 1b: sizes + convert W -> bf16 in the same streaming pass
// ---------------------------------------------------------------------------
__global__ __launch_bounds__(256) void pcl_sizes_conv(const float* __restrict__ W,
                                                      __bf16* __restrict__ Wbf,
                                                      float* __restrict__ invsz,
                                                      float* __restrict__ validf) {
    const int p = blockIdx.x;
    __shared__ float red[256];
    const float*  row = W   + (size_t)p * NGENES;
    __bf16*      orow = Wbf + (size_t)p * NGENES;
    float s = 0.0f;
    for (int g = threadIdx.x * 8; g < NGENES; g += 256 * 8) {
        float4 a = *(const float4*)(row + g);
        float4 b = *(const float4*)(row + g + 4);
        s += (a.x + a.y) + (a.z + a.w) + (b.x + b.y) + (b.z + b.w);
        v8bf o;
        o[0] = (__bf16)a.x; o[1] = (__bf16)a.y; o[2] = (__bf16)a.z; o[3] = (__bf16)a.w;
        o[4] = (__bf16)b.x; o[5] = (__bf16)b.y; o[6] = (__bf16)b.z; o[7] = (__bf16)b.w;
        *(uint4*)(orow + g) = __builtin_bit_cast(uint4, o);
    }
    red[threadIdx.x] = s;
    __syncthreads();
    for (int off = 128; off > 0; off >>= 1) {
        if (threadIdx.x < off) red[threadIdx.x] += red[threadIdx.x + off];
        __syncthreads();
    }
    if (threadIdx.x == 0) {
        float sz = red[0];
        bool  v  = (sz >= 5.0f);
        float ss = fmaxf(sz, 1.0f);
        validf[p] = v ? 1.0f : 0.0f;
        invsz[p]  = v ? 1.0f / (ss * ss) : 0.0f;
    }
}

// ---------------------------------------------------------------------------
// Fragment helpers (wave32 VGPR layouts, see CDNA5 ISA)
// ---------------------------------------------------------------------------
__device__ __forceinline__ v16bf frag_from_2x16B(const __bf16* lo16, const __bf16* hi16) {
    v8bf lo = __builtin_bit_cast(v8bf, *(const uint4*)lo16);
    v8bf hh = __builtin_bit_cast(v8bf, *(const uint4*)hi16);
    return __builtin_shufflevector(lo, hh, 0, 1, 2, 3, 4, 5, 6, 7,
                                           8, 9, 10, 11, 12, 13, 14, 15);
}

__device__ __forceinline__ v16bf load_a_frag(const __bf16* __restrict__ drow,
                                             int k0, int hi) {
    return frag_from_2x16B(drow + k0 + hi * 16, drow + k0 + hi * 16 + 8);
}

__device__ __forceinline__ v16bf load_b_frag_f32(const float* __restrict__ wrow,
                                                 int k0, int hi) {
    const float* wr = wrow + k0 + hi * 16;
    float4 w0 = *(const float4*)(wr);
    float4 w1 = *(const float4*)(wr + 4);
    float4 w2 = *(const float4*)(wr + 8);
    float4 w3 = *(const float4*)(wr + 12);
    v16bf b;
    b[0]  = (__bf16)w0.x; b[1]  = (__bf16)w0.y; b[2]  = (__bf16)w0.z; b[3]  = (__bf16)w0.w;
    b[4]  = (__bf16)w1.x; b[5]  = (__bf16)w1.y; b[6]  = (__bf16)w1.z; b[7]  = (__bf16)w1.w;
    b[8]  = (__bf16)w2.x; b[9]  = (__bf16)w2.y; b[10] = (__bf16)w2.z; b[11] = (__bf16)w2.w;
    b[12] = (__bf16)w3.x; b[13] = (__bf16)w3.y; b[14] = (__bf16)w3.z; b[15] = (__bf16)w3.w;
    return b;
}

// ---------------------------------------------------------------------------
// Kernel 2 (big path): LDS-staged WMMA GEMM with GLOBAL_LOAD_ASYNC_TO_LDS.
//
// Block: 256 threads = 8 waves (4 M x 2 N). Block tile 64(M) x 128(N), K=32.
// LDS stage: A tile 64x32 bf16 (4 KB) + B tile 128x32 bf16 (8 KB) = 12 KB;
// triple-buffered (36 KB). Fill = 3 async b128 ops per thread; pipeline:
//   wait(asynccnt<=3) -> barrier -> issue fill(i+2) ->
//   issue ALL ds_loads (A + 4xB) -> 4 back-to-back WMMAs
// The barrier both publishes stage i and proves stage (i+2)%3's readers done.
// ---------------------------------------------------------------------------
#define STAGE_BYTES 12288            // 4096 (A) + 8192 (B)
#define STAGE_ELEMS 6144             // in bf16

__global__ __launch_bounds__(256) void pcl_gemm_lds(const __bf16* __restrict__ Dbf,
                                                    const __bf16* __restrict__ Wbf,
                                                    float* __restrict__ Mout) {
    __shared__ __bf16 ldsbuf[3 * STAGE_ELEMS];   // 36 KB

    const int t      = threadIdx.x;
    const int lane   = t & 31;
    const int wave   = t >> 5;
    const int wave_m = wave & 3;          // 0..3
    const int wave_n = wave >> 2;         // 0..1
    const int rowBase0 = blockIdx.x * 64;                 // block batch-row base
    const int colBase0 = blockIdx.y * 128;                // block pathway-col base
    const int rowBase  = rowBase0 + wave_m * 16;
    const int colBase  = colBase0 + wave_n * 64;

    const int m  = lane & 15;
    const int hi = lane >> 4;

    // Per-thread fill assignment (constant across stages)
    const int a_row = t >> 2, a_seg = t & 3;              // 256 chunks of 16B
    const size_t a_goff = (size_t)(rowBase0 + a_row) * NGENES + a_seg * 8;
    int b_col[2], b_seg[2];
    size_t b_goff[2];
#pragma unroll
    for (int j = 0; j < 2; ++j) {
        int c = t + j * 256;                               // 512 chunks of 16B
        b_col[j] = c >> 2; b_seg[j] = c & 3;
        int pc = colBase0 + b_col[j];
        if (pc > NPATH - 1) pc = NPATH - 1;
        b_goff[j] = (size_t)pc * NGENES + b_seg[j] * 8;
    }

    const unsigned lds0 = (unsigned)(size_t)&ldsbuf[0];    // LDS byte address

    v8f acc[4];
#pragma unroll
    for (int f = 0; f < 4; ++f) acc[f] = (v8f){0, 0, 0, 0, 0, 0, 0, 0};

    const int steps = NGENES / 32;                         // 625

    // Issue one stage fill: 3 async b128 loads (A:1, B:2)
    auto issue_fill = [&](int stage, int k) {
        unsigned base = lds0 + (unsigned)stage * STAGE_BYTES;
        {
            unsigned la = base + a_row * 64 + a_seg * 16;
            const __bf16* g = Dbf + a_goff + k;
            asm volatile("global_load_async_to_lds_b128 %0, %1, off"
                         :: "v"(la), "v"(g) : "memory");
        }
#pragma unroll
        for (int j = 0; j < 2; ++j) {
            unsigned lb = base + 4096 + b_col[j] * 64 + b_seg[j] * 16;
            const __bf16* g = Wbf + b_goff[j] + k;
            asm volatile("global_load_async_to_lds_b128 %0, %1, off"
                         :: "v"(lb), "v"(g) : "memory");
        }
    };

    // Prologue: fill stages 0 and 1
    issue_fill(0, 0);
    issue_fill(1, 32);

    for (int i = 0; i < steps; ++i) {
        // Wait: stage i's fill complete (stage i+1 may still be in flight)
        if (i + 1 < steps) asm volatile("s_wait_asynccnt 3" ::: "memory");
        else               asm volatile("s_wait_asynccnt 0" ::: "memory");
        __syncthreads();   // publish stage i to all waves; readers of (i+2)%3 done

        if (i + 2 < steps) issue_fill((i + 2) % 3, (i + 2) * 32);

        const __bf16* stg = &ldsbuf[(i % 3) * STAGE_ELEMS];

        // Issue ALL LDS fragment loads first (A + 4xB), then burst 4 WMMAs:
        // one dscnt drain per K-step instead of one per WMMA.
        const __bf16* arow = stg + (wave_m * 16 + m) * 32;
        v16bf a = frag_from_2x16B(arow + hi * 16, arow + hi * 16 + 8);

        v16bf b0, b1, b2, b3;
        {
            const __bf16* c0 = stg + 2048 + (wave_n * 64 +  0 + m) * 32;
            const __bf16* c1 = stg + 2048 + (wave_n * 64 + 16 + m) * 32;
            const __bf16* c2 = stg + 2048 + (wave_n * 64 + 32 + m) * 32;
            const __bf16* c3 = stg + 2048 + (wave_n * 64 + 48 + m) * 32;
            b0 = frag_from_2x16B(c0 + hi * 16, c0 + hi * 16 + 8);
            b1 = frag_from_2x16B(c1 + hi * 16, c1 + hi * 16 + 8);
            b2 = frag_from_2x16B(c2 + hi * 16, c2 + hi * 16 + 8);
            b3 = frag_from_2x16B(c3 + hi * 16, c3 + hi * 16 + 8);
        }

        acc[0] = __builtin_amdgcn_wmma_f32_16x16x32_bf16(
            false, a, false, b0, (short)0, acc[0], false, false);
        acc[1] = __builtin_amdgcn_wmma_f32_16x16x32_bf16(
            false, a, false, b1, (short)0, acc[1], false, false);
        acc[2] = __builtin_amdgcn_wmma_f32_16x16x32_bf16(
            false, a, false, b2, (short)0, acc[2], false, false);
        acc[3] = __builtin_amdgcn_wmma_f32_16x16x32_bf16(
            false, a, false, b3, (short)0, acc[3], false, false);
    }

    // Store D: VGPR r -> row (rowBase + hi*8 + r), lane m -> column
#pragma unroll
    for (int f = 0; f < 4; ++f) {
        int col = colBase + f * 16 + m;
        if (col < NPATH) {
#pragma unroll
            for (int r = 0; r < 8; ++r) {
                int row = rowBase + hi * 8 + r;
                Mout[(size_t)row * NPATH + col] = acc[f][r];
            }
        }
    }
}

// ---------------------------------------------------------------------------
// Kernel 2 (fallback, small workspace): direct-global WMMA GEMM, W in f32
// ---------------------------------------------------------------------------
__global__ __launch_bounds__(256) void pcl_gemm_direct(const __bf16* __restrict__ Dbf,
                                                       const float* __restrict__ W,
                                                       float* __restrict__ Mout) {
    const int lane   = threadIdx.x & 31;
    const int wave   = threadIdx.x >> 5;
    const int wave_m = wave & 3;
    const int wave_n = wave >> 2;
    const int rowBase = blockIdx.x * 64  + wave_m * 16;
    const int colBase = blockIdx.y * 128 + wave_n * 64;

    const int m  = lane & 15;
    const int hi = lane >> 4;

    v8f acc[4];
#pragma unroll
    for (int f = 0; f < 4; ++f) acc[f] = (v8f){0, 0, 0, 0, 0, 0, 0, 0};

    const __bf16* drow = Dbf + (size_t)(rowBase + m) * NGENES;
    const float* wrow[4];
#pragma unroll
    for (int f = 0; f < 4; ++f) {
        int c = colBase + f * 16 + m;
        if (c > NPATH - 1) c = NPATH - 1;
        wrow[f] = W + (size_t)c * NGENES;
    }

    for (int k0 = 0; k0 < NGENES; k0 += 32) {
        if (k0 + 256 < NGENES) {
            __builtin_prefetch(drow + k0 + 256, 0, 1);
            __builtin_prefetch(wrow[0] + k0 + 256, 0, 1);
        }
        // Load everything first, then burst the 4 WMMAs.
        v16bf a  = load_a_frag(drow, k0, hi);
        v16bf b0 = load_b_frag_f32(wrow[0], k0, hi);
        v16bf b1 = load_b_frag_f32(wrow[1], k0, hi);
        v16bf b2 = load_b_frag_f32(wrow[2], k0, hi);
        v16bf b3 = load_b_frag_f32(wrow[3], k0, hi);
        acc[0] = __builtin_amdgcn_wmma_f32_16x16x32_bf16(
            false, a, false, b0, (short)0, acc[0], false, false);
        acc[1] = __builtin_amdgcn_wmma_f32_16x16x32_bf16(
            false, a, false, b1, (short)0, acc[1], false, false);
        acc[2] = __builtin_amdgcn_wmma_f32_16x16x32_bf16(
            false, a, false, b2, (short)0, acc[2], false, false);
        acc[3] = __builtin_amdgcn_wmma_f32_16x16x32_bf16(
            false, a, false, b3, (short)0, acc[3], false, false);
    }

#pragma unroll
    for (int f = 0; f < 4; ++f) {
        int col = colBase + f * 16 + m;
        if (col < NPATH) {
#pragma unroll
            for (int r = 0; r < 8; ++r) {
                int row = rowBase + hi * 8 + r;
                Mout[(size_t)row * NPATH + col] = acc[f][r];
            }
        }
    }
}

// ---------------------------------------------------------------------------
// Kernel 3: contrib[p] = invsz[p] * mean_b( M[b,p]^2 )
// ---------------------------------------------------------------------------
__global__ __launch_bounds__(256) void pcl_pathred(const float* __restrict__ Mout,
                                                   const float* __restrict__ invsz,
                                                   float* __restrict__ contrib) {
    const int p = blockIdx.x;
    __shared__ float red[256];
    float v = Mout[(size_t)threadIdx.x * NPATH + p];
    red[threadIdx.x] = v * v;
    __syncthreads();
    for (int off = 128; off > 0; off >>= 1) {
        if (threadIdx.x < off) red[threadIdx.x] += red[threadIdx.x + off];
        __syncthreads();
    }
    if (threadIdx.x == 0)
        contrib[p] = red[0] * invsz[p] * (1.0f / (float)BATCH);
}

// ---------------------------------------------------------------------------
// Kernel 4: final mean over valid pathways
// ---------------------------------------------------------------------------
__global__ __launch_bounds__(256) void pcl_final(const float* __restrict__ contrib,
                                                 const float* __restrict__ validf,
                                                 float* __restrict__ out) {
    __shared__ float redT[256];
    __shared__ float redN[256];
    float t = 0.0f, n = 0.0f;
    for (int p = threadIdx.x; p < NPATH; p += 256) {
        t += contrib[p];
        n += validf[p];
    }
    redT[threadIdx.x] = t;
    redN[threadIdx.x] = n;
    __syncthreads();
    for (int off = 128; off > 0; off >>= 1) {
        if (threadIdx.x < off) {
            redT[threadIdx.x] += redT[threadIdx.x + off];
            redN[threadIdx.x] += redN[threadIdx.x + off];
        }
        __syncthreads();
    }
    if (threadIdx.x == 0) {
        float total = redT[0], nv = redN[0];
        out[0] = (nv > 0.0f) ? total / fmaxf(nv, 1.0f) : 0.0f;
    }
}

// ---------------------------------------------------------------------------
extern "C" void kernel_launch(void* const* d_in, const int* in_sizes, int n_in,
                              void* d_out, int out_size, void* d_ws, size_t ws_size,
                              hipStream_t stream) {
    const float* expr = (const float*)d_in[0];   // (B, G)
    const float* pred = (const float*)d_in[1];   // (B, G)
    const float* W    = (const float*)d_in[2];   // (P, G)
    float*       out  = (float*)d_out;

    // Workspace layout: invsz[P] | validf[P] | contrib[P] | M[B*P] | Dbf | (Wbf)
    float* invsz   = (float*)d_ws;
    float* validf  = invsz  + NPATH;
    float* contrib = validf + NPATH;
    float* Mout    = contrib + NPATH;
    __bf16* Dbf    = (__bf16*)(Mout + (size_t)BATCH * NPATH);   // 10.24 MB
    __bf16* Wbf    = Dbf + (size_t)BATCH * NGENES;              // +80 MB (optional)

    size_t need_big = (size_t)((char*)(Wbf + (size_t)NPATH * NGENES) - (char*)d_ws);
    bool big = (ws_size >= need_big);   // ws_size is fixed -> deterministic

    // D = pred - expr, converted once to bf16
    pcl_diff<<<(BATCH * NGENES) / (256 * 8), 256, 0, stream>>>(expr, pred, Dbf);

    dim3 grid(BATCH / 64, (NPATH + 127) / 128);   // (4, 16)
    if (big) {
        pcl_sizes_conv<<<NPATH, 256, 0, stream>>>(W, Wbf, invsz, validf);
        pcl_gemm_lds<<<grid, 256, 0, stream>>>(Dbf, Wbf, Mout);
    } else {
        pcl_sizes<<<NPATH, 256, 0, stream>>>(W, invsz, validf);
        pcl_gemm_direct<<<grid, 256, 0, stream>>>(Dbf, W, Mout);
    }

    pcl_pathred<<<NPATH, 256, 0, stream>>>(Mout, invsz, contrib);
    pcl_final<<<1, 256, 0, stream>>>(contrib, validf, out);
}
